// GIN_38216619000492
// MI455X (gfx1250) — compile-verified, matrix operand order checked by reading, not run on hardware
//
#include <hip/hip_runtime.h>
#include <hip/hip_bf16.h>

// ---------------------------------------------------------------------------
// GIN layer on MI455X (gfx1250, wave32, WMMA):
//   agg  = x + segment_sum(x[src], dst)          (L2-atomic bound)
//   h    = relu(agg @ W1 + b1)                   (WMMA bf16, f32 accum)
//   out  = h @ W2 + b2                           (WMMA bf16, f32 accum)
// Wide N-tiles per wave amortize A-operand reads (4 WMMAs per A load).
// ---------------------------------------------------------------------------

typedef __attribute__((ext_vector_type(16))) __bf16 v16bf;
typedef __attribute__((ext_vector_type(8)))  __bf16 v8bf;
typedef __attribute__((ext_vector_type(8)))  float  v8f;
typedef __attribute__((ext_vector_type(4)))  float  v4f;

#define D_IN   64
#define D_H    256
#define D_OUT  64

// ---------------------------------------------------------------------------
// 1) agg = x   (eps == 0, so (1+eps)*x == x; scatter adds on top of this)
// ---------------------------------------------------------------------------
__global__ void gin_init_agg(const float* __restrict__ x,
                             float* __restrict__ agg, int total4) {
    int i = blockIdx.x * blockDim.x + threadIdx.x;
    if (i < total4) {
        ((v4f*)agg)[i] = ((const v4f*)x)[i];
    }
}

// ---------------------------------------------------------------------------
// 2) Edge scatter: one wave per edge, 2 floats per lane (float2 load),
//    relaxed agent-scope f32 atomic add -> global_atomic_add_f32 (no return).
// ---------------------------------------------------------------------------
__global__ void gin_scatter(const float* __restrict__ x,
                            const int* __restrict__ src,
                            const int* __restrict__ dst,
                            float* __restrict__ agg, int E) {
    int gid  = blockIdx.x * blockDim.x + threadIdx.x;
    int e    = gid >> 5;
    int lane = gid & 31;
    if (e >= E) return;
    int s = src[e];
    int d = dst[e];
    const float2 v = ((const float2*)x)[(size_t)s * (D_IN / 2) + lane];
    float* a = agg + (size_t)d * D_IN + lane * 2;
    __hip_atomic_fetch_add(a + 0, v.x, __ATOMIC_RELAXED, __HIP_MEMORY_SCOPE_AGENT);
    __hip_atomic_fetch_add(a + 1, v.y, __ATOMIC_RELAXED, __HIP_MEMORY_SCOPE_AGENT);
}

// ---------------------------------------------------------------------------
// 3) Repack weight matrix W[K x Ncols] (row-major f32) into bf16 tiles laid
//    out exactly as the wave32 WMMA B-operand expects, so the GEMM inner
//    loop loads one contiguous 32B v16bf per lane.
//    Tile (kt, nt) covers K rows [kt*32, kt*32+32), cols [nt*16, nt*16+16).
//    Lane l (<16): col = nt*16+l,     K chunks [0..7]  and [16..23] of tile
//    Lane l (>=16): col = nt*16+l-16, K chunks [8..15] and [24..31]
// ---------------------------------------------------------------------------
__global__ void gin_pack_b(const float* __restrict__ W, __bf16* __restrict__ P,
                           int ncols, int ntiles_n, int total) {
    int t = blockIdx.x * blockDim.x + threadIdx.x;
    if (t >= total) return;
    int lane = t & 31;
    int tile = t >> 5;
    int kt = tile / ntiles_n;
    int nt = tile - kt * ntiles_n;
    int n  = nt * 16 + (lane & 15);
    int kb = kt * 32 + ((lane >> 4) << 3);
    __bf16* o = P + (size_t)tile * 512 + lane * 16;
#pragma unroll
    for (int j = 0; j < 8; ++j) {
        o[j]     = (__bf16)W[(size_t)(kb + j)      * ncols + n];
        o[8 + j] = (__bf16)W[(size_t)(kb + 16 + j) * ncols + n];
    }
}

// ---------------------------------------------------------------------------
// 4) GEMM1: h = relu(agg @ W1 + b1).  M=N_nodes, K=64, N=256.
//    One wave per 16x64 output tile: A loaded once per k-step (f32->bf16
//    in-register) and reused by 4 back-to-back WMMAs into 4 accumulators.
//    Output stored bf16 row-major (feeds GEMM2's A loads).
// ---------------------------------------------------------------------------
__global__ void gin_gemm1(const float* __restrict__ h0,
                          const __bf16* __restrict__ W1p,
                          const float* __restrict__ b1,
                          __bf16* __restrict__ h, int nwaves) {
    int wave = blockIdx.x * (blockDim.x >> 5) + (threadIdx.x >> 5);
    if (wave >= nwaves) return;            // whole wave exits together
    int lane = threadIdx.x & 31;
    int mt = wave >> 2;                    // 4 N-blocks of 64 cols
    int nb = wave & 3;
    int m0 = mt * 16;

    int arow = m0 + (lane & 15);
    int kb   = (lane >> 4) << 3;

    v8f acc[4] = {};
#pragma unroll
    for (int kt = 0; kt < D_IN / 32; ++kt) {
        const float* ap = h0 + (size_t)arow * D_IN + kt * 32 + kb;
        v16bf a;
#pragma unroll
        for (int j = 0; j < 8; ++j) {
            a[j]     = (__bf16)ap[j];
            a[8 + j] = (__bf16)ap[16 + j];
        }
        const __bf16* bbase = W1p + ((size_t)(kt * 16 + nb * 4) * 512 + lane * 16);
#pragma unroll
        for (int t = 0; t < 4; ++t) {
            const v16bf b = *(const v16bf*)(bbase + (size_t)t * 512);
            acc[t] = __builtin_amdgcn_wmma_f32_16x16x32_bf16(
                /*neg_a=*/false, a, /*neg_b=*/false, b,
                /*c_mod=*/(short)0, acc[t], /*reuse_a=*/false, /*reuse_b=*/false);
        }
    }

    int coll  = lane & 15;
    int rbase = m0 + ((lane >> 4) << 3);   // lanes 0-15 -> rows r, 16-31 -> r+8
#pragma unroll
    for (int t = 0; t < 4; ++t) {
        int col    = (nb * 4 + t) * 16 + coll;
        float bias = b1[col];
#pragma unroll
        for (int r = 0; r < 8; ++r) {
            float v = acc[t][r] + bias;
            v = v > 0.0f ? v : 0.0f;
            h[(size_t)(rbase + r) * D_H + col] = (__bf16)v;
        }
    }
}

// ---------------------------------------------------------------------------
// 5) GEMM2: out = h @ W2 + b2.  M=N_nodes, K=256, N=64.
//    One wave per 16x64 output block (all of N): A (bf16 h rows) is loaded
//    exactly once; 8 k-steps x 4 WMMAs = 32 WMMAs per wave.
// ---------------------------------------------------------------------------
__global__ void gin_gemm2(const __bf16* __restrict__ h,
                          const __bf16* __restrict__ W2p,
                          const float* __restrict__ b2,
                          float* __restrict__ out, int nwaves) {
    int wave = blockIdx.x * (blockDim.x >> 5) + (threadIdx.x >> 5);
    if (wave >= nwaves) return;            // whole wave exits together
    int lane = threadIdx.x & 31;
    int m0 = wave * 16;

    int arow = m0 + (lane & 15);
    int kb   = (lane >> 4) << 3;

    v8f acc[4] = {};
#pragma unroll
    for (int kt = 0; kt < D_H / 32; ++kt) {
        const __bf16* ap = h + (size_t)arow * D_H + kt * 32 + kb;
        const v8bf lo = *(const v8bf*)ap;
        const v8bf hi = *(const v8bf*)(ap + 16);
        v16bf a;
#pragma unroll
        for (int j = 0; j < 8; ++j) {
            a[j]     = lo[j];
            a[8 + j] = hi[j];
        }
        const __bf16* bbase = W2p + ((size_t)(kt * 4) * 512 + lane * 16);
#pragma unroll
        for (int t = 0; t < 4; ++t) {
            const v16bf b = *(const v16bf*)(bbase + (size_t)t * 512);
            acc[t] = __builtin_amdgcn_wmma_f32_16x16x32_bf16(
                false, a, false, b, (short)0, acc[t], false, false);
        }
    }

    int coll  = lane & 15;
    int rbase = m0 + ((lane >> 4) << 3);
#pragma unroll
    for (int t = 0; t < 4; ++t) {
        int col    = t * 16 + coll;
        float bias = b2[col];
#pragma unroll
        for (int r = 0; r < 8; ++r) {
            out[(size_t)(rbase + r) * D_OUT + col] = acc[t][r] + bias;
        }
    }
}

// ---------------------------------------------------------------------------
// Launcher
// ---------------------------------------------------------------------------
extern "C" void kernel_launch(void* const* d_in, const int* in_sizes, int n_in,
                              void* d_out, int out_size, void* d_ws, size_t ws_size,
                              hipStream_t stream) {
    const float* x  = (const float*)d_in[0];
    const int*   ei = (const int*)d_in[1];   // edge_index [2, E]
    const float* W1 = (const float*)d_in[2];
    const float* b1 = (const float*)d_in[3];
    const float* W2 = (const float*)d_in[4];
    const float* b2 = (const float*)d_in[5];
    float* out = (float*)d_out;

    const int N = in_sizes[0] / D_IN;        // 50000 (multiple of 16)
    const int E = in_sizes[1] / 2;           // 800000
    const int* src = ei;
    const int* dst = ei + E;

    // Workspace carve-up (256B aligned): agg f32 | h bf16 | W1 packed | W2 packed
    char* ws = (char*)d_ws;
    size_t off = 0;
    float* agg = (float*)(ws + off);
    off += (size_t)N * D_IN * sizeof(float);           off = (off + 255) & ~(size_t)255;
    __bf16* hbuf = (__bf16*)(ws + off);
    off += (size_t)N * D_H * sizeof(__bf16);           off = (off + 255) & ~(size_t)255;
    __bf16* W1p = (__bf16*)(ws + off);
    off += (size_t)(D_IN / 32) * (D_H / 16) * 512 * sizeof(__bf16);
    off = (off + 255) & ~(size_t)255;
    __bf16* W2p = (__bf16*)(ws + off);

    // 1) agg = x
    {
        int total4 = N * D_IN / 4;
        gin_init_agg<<<(total4 + 255) / 256, 256, 0, stream>>>(x, agg, total4);
    }
    // 2) pack W1, W2 into WMMA B-operand layout (tiny, L2-resident)
    {
        int t1 = (D_IN / 32) * (D_H / 16) * 32;    // 1024
        gin_pack_b<<<(t1 + 255) / 256, 256, 0, stream>>>(W1, W1p, D_H, D_H / 16, t1);
        int t2 = (D_H / 32) * (D_OUT / 16) * 32;   // 1024
        gin_pack_b<<<(t2 + 255) / 256, 256, 0, stream>>>(W2, W2p, D_OUT, D_OUT / 16, t2);
    }
    // 3) edge scatter-add (wave per edge)
    {
        long long threads = (long long)E * 32;
        int blocks = (int)((threads + 255) / 256);
        gin_scatter<<<blocks, 256, 0, stream>>>(x, src, dst, agg, E);
    }
    // 4) GEMM1 + bias + ReLU  (wave per 16x64 tile)
    {
        int nwaves = (N / 16) * (D_H / 64);        // 12500
        gin_gemm1<<<(nwaves + 7) / 8, 256, 0, stream>>>(agg, W1p, b1, hbuf, nwaves);
    }
    // 5) GEMM2 + bias (wave per 16x64 block = full N)
    {
        int nwaves = N / 16;                       // 3125
        gin_gemm2<<<(nwaves + 7) / 8, 256, 0, stream>>>(hbuf, W2p, b2, out, nwaves);
    }
}
